// TorchFlowLenia_20839181320908
// MI455X (gfx1250) — compile-verified, matrix operand order checked by reading, not run on hardware
//
#include <hip/hip_runtime.h>

typedef __attribute__((ext_vector_type(16))) _Float16 v16h;
typedef __attribute__((ext_vector_type(8)))  float    v8f;

#define SXX   256
#define SYY   256
#define CCH   3
#define NKK   12
#define NBB   4
#define KWIN  64      // truncated kernel support window (offsets -32..31)
#define KHALO 32
#define HAL_R 79      // 16 + 63
#define HAL_C 80      // 16 + 64
#define RDD   5
#define TRR   26      // 16 + 2*RDD

// ---------------------------------------------------------------------------
// Kernel 1: pack Lenia kernels into WMMA B-fragment order, f16.
// Layout index = ((((c*KWIN + dx)*2 + kc)*32 + lane)*16 + i)
//   slot n = lane&15, dy = kc*32 + (lane>>4)*16 + i
//   value  = (n<12 && c0[n]==c) ? Kn[96+dx][96+dy][n] : 0
// ---------------------------------------------------------------------------
__global__ __launch_bounds__(256) void lenia_pack(const float* __restrict__ Kn,
                                                  const int* __restrict__ c0,
                                                  _Float16* __restrict__ Wf)
{
    int t = blockIdx.x * 256 + threadIdx.x;           // 3*64*2*32*16 = 196608
    if (t >= CCH * KWIN * 2 * 32 * 16) return;
    int i    = t & 15;
    int lane = (t >> 4) & 31;
    int kc   = (t >> 9) & 1;
    int dxi  = (t >> 10) & 63;
    int c    = t >> 16;
    int n    = lane & 15;
    int dyi  = kc * 32 + ((lane >> 4) << 4) + i;
    float v = 0.0f;
    if (n < NKK && c0[n] == c) {
        int gx = 96 + dxi;                            // 128 + (dxi-32)
        int gy = 96 + dyi;
        v = Kn[((size_t)gx * SYY + gy) * NKK + n];
    }
    Wf[t] = (_Float16)v;
}

// ---------------------------------------------------------------------------
// Kernel 2: truncated circular convolution as WMMA GEMM.
// Workgroup = 16x16 pixel patch of one batch. Each wave owns 2 pixel rows
// (M = 16 pixels along y, N = 16 kernel slots, K = 3*64*64 taps).
// A-halo staged in LDS as f16.
// ---------------------------------------------------------------------------
__global__ __launch_bounds__(256) void lenia_conv(const float* __restrict__ A,
                                                  const _Float16* __restrict__ Wf,
                                                  float* __restrict__ U)
{
    __shared__ _Float16 hal[CCH][HAL_R][HAL_C];

    const int tid = threadIdx.x;
    const int x0  = blockIdx.x * 16;
    const int y0  = blockIdx.y * 16;
    const int b   = blockIdx.z;
    const float* Ab = A + (size_t)b * SXX * SYY * CCH;

    // cooperative halo load (torus wrap), fp32 -> fp16
    for (int idx = tid; idx < CCH * HAL_R * HAL_C; idx += 256) {
        int c   = idx / (HAL_R * HAL_C);
        int rem = idx % (HAL_R * HAL_C);
        int r   = rem / HAL_C;
        int col = rem % HAL_C;
        int gx  = (x0 - KHALO + r)   & (SXX - 1);
        int gy  = (y0 - KHALO + col) & (SYY - 1);
        hal[c][r][col] = (_Float16)Ab[((size_t)gx * SYY + gy) * CCH + c];
    }
    __syncthreads();

    const int wave = tid >> 5;
    const int lane = tid & 31;
    const int m    = lane & 15;            // pixel index within M-tile (y)
    const int h16  = (lane >> 4) << 4;     // K-half select: 0 or 16
    const int xr0  = wave * 2;             // local pixel rows of this wave
    const int xr1  = wave * 2 + 1;

    v8f acc0 = {};
    v8f acc1 = {};

    for (int c = 0; c < CCH; ++c) {
        for (int dxi = 0; dxi < KWIN; ++dxi) {
            const _Float16* wrow = Wf + (((size_t)c * KWIN + dxi) << 10); // *2*32*16
            __builtin_prefetch(wrow + 2048, 0, 1);   // global_prefetch next weight rows
#pragma unroll
            for (int kc = 0; kc < 2; ++kc) {
                // B fragment: one aligned 32B load per lane (fragment-order packed)
                v16h bfrag = *(const v16h*)(wrow + ((kc * 32 + lane) << 4));
                // A fragments from LDS (per-lane 16 contiguous halves)
                const int colb = m + kc * 32 + h16;
                const _Float16* p0 = &hal[c][xr0 + dxi][colb];
                const _Float16* p1 = &hal[c][xr1 + dxi][colb];
                v16h a0, a1;
#pragma unroll
                for (int i = 0; i < 16; ++i) { a0[i] = p0[i]; a1[i] = p1[i]; }
                acc0 = __builtin_amdgcn_wmma_f32_16x16x32_f16(
                           false, a0, false, bfrag, (short)0, acc0, false, false);
                acc1 = __builtin_amdgcn_wmma_f32_16x16x32_f16(
                           false, a1, false, bfrag, (short)0, acc1, false, false);
            }
        }
    }

    // store U[b][x][y][slot] per C/D layout: VGPR r -> M=r (lanes 0-15), M=r+8 (16-31)
    const int n    = lane & 15;
    const int yoff = (lane >> 4) * 8;
    const size_t base = (size_t)b * SXX * SYY * 16;
#pragma unroll
    for (int r = 0; r < 8; ++r) {
        int y = y0 + r + yoff;
        U[base + (((size_t)(x0 + xr0) * SYY + y) << 4) + n] = acc0[r];
        U[base + (((size_t)(x0 + xr1) * SYY + y) << 4) + n] = acc1[r];
    }
}

// ---------------------------------------------------------------------------
// Kernel 3: growth + sum into target channels: Uc[b,x,y,c]
// ---------------------------------------------------------------------------
__global__ __launch_bounds__(256) void lenia_growth(const float* __restrict__ U,
                                                    const float* __restrict__ mk,
                                                    const float* __restrict__ sk,
                                                    const float* __restrict__ hk,
                                                    const int* __restrict__ c1,
                                                    float* __restrict__ Uc)
{
    int p = blockIdx.x * 256 + threadIdx.x;
    if (p >= NBB * SXX * SYY) return;
    const float* u = U + (size_t)p * 16;
    float o0 = 0.f, o1 = 0.f, o2 = 0.f;
#pragma unroll
    for (int k = 0; k < NKK; ++k) {
        float d  = u[k] - mk[k];
        float sv = sk[k];
        float g  = (2.0f * __expf(-(d * d) / (2.0f * sv * sv)) - 1.0f) * hk[k];
        int cc = c1[k];
        o0 += (cc == 0) ? g : 0.0f;
        o1 += (cc == 1) ? g : 0.0f;
        o2 += (cc == 2) ? g : 0.0f;
    }
    float* o = Uc + (size_t)p * 3;
    o[0] = o0; o[1] = o1; o[2] = o2;
}

// ---------------------------------------------------------------------------
// Kernel 4: Sobel(Uc), Sobel(sum A), alpha, flow -> target centers mus[b,x,y,2,c]
// ---------------------------------------------------------------------------
__global__ __launch_bounds__(256) void lenia_flow(const float* __restrict__ A,
                                                  const float* __restrict__ Uc,
                                                  float* __restrict__ mus)
{
    int p = blockIdx.x * 256 + threadIdx.x;
    if (p >= NBB * SXX * SYY) return;
    int b  = p / (SXX * SYY);
    int xy = p % (SXX * SYY);
    int x  = xy / SYY;
    int y  = xy % SYY;

    float uc[3][3][3];
    float as[3][3];
#pragma unroll
    for (int di = -1; di <= 1; ++di) {
#pragma unroll
        for (int dj = -1; dj <= 1; ++dj) {
            int xi = (x + di) & (SXX - 1);
            int yj = (y + dj) & (SYY - 1);
            size_t q = ((size_t)b * SXX + xi) * SYY + yj;
            const float* ap = A  + q * 3;
            const float* up = Uc + q * 3;
            as[di + 1][dj + 1] = ap[0] + ap[1] + ap[2];
#pragma unroll
            for (int c = 0; c < 3; ++c) uc[di + 1][dj + 1][c] = up[c];
        }
    }
    // Sobel of sum(A)
    float gxA = ((as[2][0] + 2.f * as[2][1] + as[2][2]) -
                 (as[0][0] + 2.f * as[0][1] + as[0][2])) * 0.125f;
    float gyA = ((as[0][2] + 2.f * as[1][2] + as[2][2]) -
                 (as[0][0] + 2.f * as[1][0] + as[2][0])) * 0.125f;

    size_t qc = ((size_t)b * SXX + x) * SYY + y;
    const float* ac = A + qc * 3;
    float* mo = mus + qc * 6;            // [2][C]
#pragma unroll
    for (int c = 0; c < 3; ++c) {
        float gxU = ((uc[2][0][c] + 2.f * uc[2][1][c] + uc[2][2][c]) -
                     (uc[0][0][c] + 2.f * uc[0][1][c] + uc[0][2][c])) * 0.125f;
        float gyU = ((uc[0][2][c] + 2.f * uc[1][2][c] + uc[2][2][c]) -
                     (uc[0][0][c] + 2.f * uc[1][0][c] + uc[2][0][c])) * 0.125f;
        float a     = ac[c];
        float alpha = fminf(a * a, 1.0f);        // (A/theta)^2 clipped, A>=0
        float Fx = gxU * (1.0f - alpha) - gxA * alpha;
        float Fy = gyU * (1.0f - alpha) - gyA * alpha;
        float cx = fminf(fmaxf(0.2f * Fx, -4.35f), 4.35f);   // DT*F clipped to DD-SIGMA
        float cy = fminf(fmaxf(0.2f * Fy, -4.35f), 4.35f);
        mo[c]     = (float)x + 0.5f + cx;
        mo[3 + c] = (float)y + 0.5f + cy;
    }
}

// ---------------------------------------------------------------------------
// Kernel 5: reintegration tracking over (2*DD+1)^2 = 121 offsets, LDS-tiled.
// ---------------------------------------------------------------------------
__global__ __launch_bounds__(256) void lenia_rt(const float* __restrict__ A,
                                                const float* __restrict__ mus,
                                                float* __restrict__ nA)
{
    __shared__ float sA [TRR][TRR][3];
    __shared__ float sMu[TRR][TRR][6];

    const int tid = threadIdx.x;
    const int x0  = blockIdx.x * 16;
    const int y0  = blockIdx.y * 16;
    const int b   = blockIdx.z;

    for (int idx = tid; idx < TRR * TRR; idx += 256) {
        int r   = idx / TRR;
        int col = idx % TRR;
        int gx = (x0 - RDD + r)   & (SXX - 1);
        int gy = (y0 - RDD + col) & (SYY - 1);
        size_t q = ((size_t)b * SXX + gx) * SYY + gy;
        const float* ap = A   + q * 3;
        const float* mp = mus + q * 6;
        sA[r][col][0] = ap[0]; sA[r][col][1] = ap[1]; sA[r][col][2] = ap[2];
#pragma unroll
        for (int w = 0; w < 6; ++w) sMu[r][col][w] = mp[w];
    }
    __syncthreads();

    const int px = tid >> 4;
    const int py = tid & 15;
    const float posx = (float)(x0 + px) + 0.5f;
    const float posy = (float)(y0 + py) + 0.5f;
    const float inv4s2 = 1.0f / (4.0f * 0.65f * 0.65f);

    float a0 = 0.f, a1 = 0.f, a2 = 0.f;
    for (int dx = -RDD; dx <= RDD; ++dx) {
        for (int dy = -RDD; dy <= RDD; ++dy) {
            int r   = px + RDD - dx;     // source pixel (x-dx, y-dy)
            int col = py + RDD - dy;
#pragma unroll
            for (int c = 0; c < 3; ++c) {
                float mux = sMu[r][col][c];
                float muy = sMu[r][col][3 + c];
                float szx = 0.5f - fabsf(posx - mux) + 0.65f;
                float szy = 0.5f - fabsf(posy - muy) + 0.65f;
                float axc = fminf(fmaxf(szx, 0.0f), 1.0f);   // amax = min(1, 2*sigma) = 1
                float ayc = fminf(fmaxf(szy, 0.0f), 1.0f);
                float area = axc * ayc * inv4s2;
                float av = sA[r][col][c];
                if (c == 0) a0 += av * area;
                else if (c == 1) a1 += av * area;
                else a2 += av * area;
            }
        }
    }
    float* o = nA + (((size_t)b * SXX + (x0 + px)) * SYY + (y0 + py)) * 3;
    o[0] = a0; o[1] = a1; o[2] = a2;
}

// ---------------------------------------------------------------------------
extern "C" void kernel_launch(void* const* d_in, const int* in_sizes, int n_in,
                              void* d_out, int out_size, void* d_ws, size_t ws_size,
                              hipStream_t stream) {
    (void)in_sizes; (void)n_in; (void)out_size; (void)ws_size;
    const float* A   = (const float*)d_in[0];
    const float* Kn  = (const float*)d_in[1];
    const float* mk  = (const float*)d_in[2];
    const float* sk  = (const float*)d_in[3];
    const float* hk  = (const float*)d_in[4];
    const int*   c0  = (const int*)d_in[5];
    const int*   c1  = (const int*)d_in[6];
    float* out = (float*)d_out;

    char* ws = (char*)d_ws;
    _Float16* Wf  = (_Float16*)ws;                               //   384 KB packed weights
    float*    U   = (float*)(ws + (512ull << 10));               //  16 MB conv output
    float*    Uc  = (float*)(ws + (512ull << 10) + (16ull << 20));
    float*    mus = (float*)(ws + (512ull << 10) + (16ull << 20) + (4ull << 20));

    const int npix = NBB * SXX * SYY;

    lenia_pack  <<<768, 256, 0, stream>>>(Kn, c0, Wf);
    lenia_conv  <<<dim3(16, 16, NBB), 256, 0, stream>>>(A, Wf, U);
    lenia_growth<<<(npix + 255) / 256, 256, 0, stream>>>(U, mk, sk, hk, c1, Uc);
    lenia_flow  <<<(npix + 255) / 256, 256, 0, stream>>>(A, Uc, mus);
    lenia_rt    <<<dim3(16, 16, NBB), 256, 0, stream>>>(A, mus, out);
}
